// SelectK_18863496364333
// MI455X (gfx1250) — matrix operand order repaired and checked by reading, not compiled, not run on hardware
//
#include <hip/hip_runtime.h>
#include <hip/hip_bf16.h>

typedef __attribute__((ext_vector_type(16))) _Float16 v16h;
typedef __attribute__((ext_vector_type(8)))  _Float16 v8h;
typedef __attribute__((ext_vector_type(8)))  float    v8f;

#define WMMA_F16(a, b, c) \
  __builtin_amdgcn_wmma_f32_16x16x32_f16(false, (a), false, (b), (short)0, (c), false, false)

// ---------------------------------------------------------------------------
// Fragment loaders (wave32; contiguous k = 16*(lane/16) + e, same order for A and B)
// ---------------------------------------------------------------------------

// A: 16x32 f16 tile from row-major f16 array (LDS or global), row stride lda
// (halfs). Two contiguous 16B loads per lane -> ds/global_load_b128.
__device__ __forceinline__ v16h load_a_frag(const _Float16* A, int lda, int m0, int k0, int lane) {
  int m  = m0 + (lane & 15);
  int kb = (lane >> 4) ? 16 : 0;
  const _Float16* p = A + (long)m * lda + k0 + kb;
  v8h lo = *(const v8h*)(p);
  v8h hi = *(const v8h*)(p + 8);
  return __builtin_shufflevector(lo, hi, 0, 1, 2, 3, 4, 5, 6, 7,
                                 8, 9, 10, 11, 12, 13, 14, 15);
}

// B: 32x16 f16 tile from pre-swizzled storage: element e of the fragment is
// k_local==16*(lane/16)+e, so each lane reads 32 contiguous bytes (coalesced across wave).
// Storage offset: (((nt*KT)+kt)*32 + lane)*16 + e.
__device__ __forceinline__ v16h load_b_swz(const _Float16* swz, int KT, int kt, int nt, int lane) {
  return *(const v16h*)(swz + (((long)nt * KT + kt) * 32 + lane) * 16);
}

// ---------------------------------------------------------------------------
// Swizzle-convert GRU weights: src f32 [3072][K] (gate-major rows, k cols),
// dst = swizzled f16 fragments with Kpad=1024 (KT=32), Npad=3072 (NT=192).
// B[k][n] = src[n][k] (the x@W^T / h@W^T transpose is folded in here).
// ---------------------------------------------------------------------------
__global__ __launch_bounds__(256) void swzT_kernel(const float* __restrict__ src,
                                                   _Float16* __restrict__ dst, int K) {
  long idx = (long)blockIdx.x * 256 + threadIdx.x;
  if (idx >= 3072L * 1024) return;
  int  j    = (int)(idx & 15);
  int  l    = (int)((idx >> 4) & 31);
  long tile = idx >> 9;                 // nt*32 + kt  (KT = 32)
  int  kt   = (int)(tile & 31);
  int  nt   = (int)(tile >> 5);
  int  n = nt * 16 + (l & 15);
  int  k = kt * 32 + (l >> 4) * 16 + j;
  dst[idx] = (k < K) ? (_Float16)src[(long)n * K + k] : (_Float16)0.f;
}

// ---------------------------------------------------------------------------
// Swizzle-convert gat_W: src f32 [300][300] already k-major (B[k][n]=src[k][n]).
// Kpad=320 (KT=10), Npad=304 (NT=19). total = 19*10*512 = 97280 halfs.
// ---------------------------------------------------------------------------
__global__ __launch_bounds__(256) void swz_gatw_kernel(const float* __restrict__ src,
                                                       _Float16* __restrict__ dst) {
  int idx = blockIdx.x * 256 + threadIdx.x;
  if (idx >= 19 * 10 * 512) return;
  int j    = idx & 15;
  int l    = (idx >> 4) & 31;
  int tile = idx >> 9;                  // nt*10 + kt (KT = 10)
  int kt   = tile % 10;
  int nt   = tile / 10;
  int n = nt * 16 + (l & 15);
  int k = kt * 32 + (l >> 4) * 16 + j;
  dst[idx] = (k < 300 && n < 300) ? (_Float16)src[(long)k * 300 + n] : (_Float16)0.f;
}

// ---------------------------------------------------------------------------
// GAT: one block per (b,s). emb gather -> WMMA (32x320)x(320x304) -> attention.
// Writes sig[s][b][0:600] = concat(word, gat_out).
// ---------------------------------------------------------------------------
__global__ __launch_bounds__(256) void gat_kernel(const int* __restrict__ xidx,
                                                  const float* __restrict__ X,
                                                  const _Float16* __restrict__ gatWswz,
                                                  const float* __restrict__ asrc,
                                                  const float* __restrict__ adst,
                                                  const float* __restrict__ gatb,
                                                  float* __restrict__ sig) {
  __shared__ alignas(32) _Float16 s_emb[32][320];
  __shared__ float s_hh[32][304];
  __shared__ float s_e[32][4];
  __shared__ float s_a[32][4];
  __shared__ float s_edst[4];

  int bs = blockIdx.x;            // 0..1023 = b*256+s
  int b = bs >> 8, s = bs & 255;
  int tid = threadIdx.x;

  _Float16* pe = &s_emb[0][0];
  for (int e = tid; e < 32 * 320; e += 256) {
    int g = e / 320, k = e % 320;
    int node = xidx[bs * 32 + g];
    pe[e] = (k < 300) ? (_Float16)X[(long)node * 300 + k] : (_Float16)0.f;
  }
  {
    int node0 = xidx[bs * 32];
    for (int k = tid; k < 300; k += 256)
      sig[(long)(s * 4 + b) * 600 + k] = X[(long)node0 * 300 + k];
  }
  __syncthreads();

  int wave = tid >> 5, lane = tid & 31;
  for (int job = wave; job < 2 * 19; job += 8) {
    int m0 = (job & 1) * 16, nt = job >> 1;
    v8f acc = {};
#pragma unroll
    for (int kt = 0; kt < 10; ++kt) {
      v16h a  = load_a_frag(&s_emb[0][0], 320, m0, kt * 32, lane);
      v16h bf = load_b_swz(gatWswz, 10, kt, nt, lane);
      acc = WMMA_F16(a, bf, acc);
    }
    int n = nt * 16 + (lane & 15);
#pragma unroll
    for (int i = 0; i < 8; ++i)
      s_hh[m0 + i + 8 * (lane >> 4)][n] = acc[i];
  }
  __syncthreads();

  if (tid < 128) {
    int g = tid >> 2, h = tid & 3;
    float a = 0.f;
    for (int d = 0; d < 75; ++d) a += s_hh[g][h * 75 + d] * asrc[h * 75 + d];
    s_e[g][h] = a;
  } else if (tid < 132) {
    int h = tid - 128;
    float a = 0.f;
    for (int d = 0; d < 75; ++d) a += s_hh[0][h * 75 + d] * adst[h * 75 + d];
    s_edst[h] = a;
  }
  __syncthreads();

  if (tid < 4) {
    int h = tid;
    float tv[32], mx = -1e30f;
    for (int g = 0; g < 32; ++g) {
      float e = s_e[g][h] + s_edst[h];
      e = (e > 0.f) ? e : 0.2f * e;   // leaky_relu 0.2
      tv[g] = e; mx = fmaxf(mx, e);
    }
    float sum = 0.f;
    for (int g = 0; g < 32; ++g) { tv[g] = __expf(tv[g] - mx); sum += tv[g]; }
    float inv = 1.f / sum;
    for (int g = 0; g < 32; ++g) s_a[g][h] = tv[g] * inv;
  }
  __syncthreads();

  for (int d = tid; d < 300; d += 256) {
    int h = d / 75;
    float a = 0.f;
    for (int g = 0; g < 32; ++g) a += s_a[g][h] * s_hh[g][d];
    sig[(long)(s * 4 + b) * 600 + 300 + d] = a + gatb[d];
  }
}

// ---------------------------------------------------------------------------
// One GRU layer: single persistent workgroup (16 waves), h kept in LDS,
// per-step WMMA [16xK]@[Kx3072] for x@Wih^T and h@Whh^T with swizzled weights.
// xin: f32 [256][4][in_dim]; out: f32 [256][4][1024];
// outTh (optional): f16 [4*256][1024], row b*256+t (B*S-major, for GEMM A).
// ---------------------------------------------------------------------------
__global__ __launch_bounds__(512) void gru_kernel(const float* __restrict__ xin, int in_dim,
                                                  const _Float16* __restrict__ WihSwz,
                                                  const _Float16* __restrict__ WhhSwz,
                                                  const float* __restrict__ bih,
                                                  const float* __restrict__ bhh,
                                                  float* __restrict__ out,
                                                  _Float16* __restrict__ outTh) {
  __shared__ alignas(32) _Float16 s_x[16][1024];
  __shared__ alignas(32) _Float16 s_h[16][1024];
  __shared__ float    s_gi[4][3072];
  __shared__ _Float16 s_gh[4][3072];

  int tid = threadIdx.x, wave = tid >> 5, lane = tid & 31;

  for (int i = tid; i < 16 * 1024; i += 512) {
    (&s_x[0][0])[i] = (_Float16)0.f;
    (&s_h[0][0])[i] = (_Float16)0.f;
  }
  __syncthreads();

  int ktx = ((in_dim + 31) & ~31) >> 5;     // K tiles for the input GEMM

  for (int t = 0; t < 256; ++t) {
    for (int i = tid; i < 4 * in_dim; i += 512) {
      int m = i / in_dim, k = i % in_dim;
      s_x[m][k] = (_Float16)xin[(long)t * 4 * in_dim + i];
    }
    __syncthreads();

    for (int j = 0; j < 12; ++j) {            // 16 waves * 12 = 192 N-tiles
      int nt = wave * 12 + j;
      v8f ai = {};
      for (int kt = 0; kt < ktx; ++kt) {
        v16h a  = load_a_frag(&s_x[0][0], 1024, 0, kt * 32, lane);
        v16h bb = load_b_swz(WihSwz, 32, kt, nt, lane);
        ai = WMMA_F16(a, bb, ai);
      }
      v8f ah = {};
      for (int kt = 0; kt < 32; ++kt) {
        v16h a  = load_a_frag(&s_h[0][0], 1024, 0, kt * 32, lane);
        v16h bb = load_b_swz(WhhSwz, 32, kt, nt, lane);
        ah = WMMA_F16(a, bb, ah);
      }
      int n = nt * 16 + (lane & 15);
      if ((lane >> 4) == 0) {                 // VGPR i -> row i; rows 0..3 valid
#pragma unroll
        for (int i = 0; i < 4; ++i) {
          s_gi[i][n] = ai[i];
          s_gh[i][n] = (_Float16)ah[i];
        }
      }
    }
    __syncthreads();

    for (int i = tid; i < 4 * 1024; i += 512) {
      int m = i >> 10, n = i & 1023;
      float gir = s_gi[m][n]        + bih[n];
      float giz = s_gi[m][n + 1024] + bih[n + 1024];
      float gin = s_gi[m][n + 2048] + bih[n + 2048];
      float ghr = (float)s_gh[m][n]        + bhh[n];
      float ghz = (float)s_gh[m][n + 1024] + bhh[n + 1024];
      float ghn = (float)s_gh[m][n + 2048] + bhh[n + 2048];
      float r  = 1.f / (1.f + __expf(-(gir + ghr)));
      float z  = 1.f / (1.f + __expf(-(giz + ghz)));
      float nn = tanhf(gin + r * ghn);
      float hold = (float)s_h[m][n];
      float hnew = (1.f - z) * nn + z * hold;
      s_h[m][n] = (_Float16)hnew;
      out[((long)t * 4 + m) * 1024 + n] = hnew;
      if (outTh) outTh[((long)m * 256 + t) * 1024 + n] = (_Float16)hnew;
    }
    __syncthreads();
  }
}

// ---------------------------------------------------------------------------
// Big projection: out[M=1024][N] = Ah[f16 1024x1024] @ W[f32 N x 1024]^T + bias.
// Block = 8 waves, tile 128(M) x 64(N). fp32 weights streamed once, converted
// to f16 and written to LDS in *fragment order* so B-fragment reads are two
// ds_load_b128 per lane. Memory-bound on W traffic (~307MB total @ 23.3TB/s).
// ---------------------------------------------------------------------------
__global__ __launch_bounds__(256) void logits_kernel(const _Float16* __restrict__ Ah,
                                                     const float* __restrict__ W,
                                                     const float* __restrict__ bias,
                                                     float* __restrict__ out, int N) {
  __shared__ alignas(32) _Float16 s_bs[4 * 32 * 16];   // 4 n-tiles, swizzled
  int tid = threadIdx.x, wave = tid >> 5, lane = tid & 31;
  int n0blk = blockIdx.x * 64;
  int m0    = blockIdx.y * 128 + wave * 16;

  v8f acc[4] = {};
  int kk = tid >> 3;            // 0..31 (k within tile)
  int jj = kk & 15;             // element within fragment
  int kb = kk >> 4;             // k half
  int nbase = (tid & 7) * 8;    // 8 consecutive n per thread

  for (int k0 = 0; k0 < 1024; k0 += 32) {
#pragma unroll
    for (int u = 0; u < 8; ++u) {
      int nl = nbase + u;
      int gn = n0blk + nl;
      int l  = (nl & 15) + 16 * kb;
      s_bs[(((nl >> 4) * 32) + l) * 16 + jj] =
          (gn < N) ? (_Float16)W[(long)gn * 1024 + k0 + kk] : (_Float16)0.f;
    }
    {   // prefetch next K-chunk of this weight row stripe (global_prefetch_b8)
      int gn = n0blk + nbase;
      if (gn < N && k0 + 32 < 1024) __builtin_prefetch(&W[(long)gn * 1024 + k0 + 32 + kk], 0, 1);
    }
    __syncthreads();
    v16h a = load_a_frag(Ah, 1024, m0, k0, lane);
#pragma unroll
    for (int j = 0; j < 4; ++j) {
      v16h bb = load_b_swz(s_bs, 1, 0, j, lane);   // KT=1, nt=j
      acc[j] = WMMA_F16(a, bb, acc[j]);
    }
    __syncthreads();
  }

  int nl = lane & 15, mrow = 8 * (lane >> 4);
#pragma unroll
  for (int j = 0; j < 4; ++j) {
    int n = n0blk + j * 16 + nl;
    if (n < N) {
      float bv = bias[n];
#pragma unroll
      for (int i = 0; i < 8; ++i)
        out[(long)(m0 + mrow + i) * N + n] = acc[j][i] + bv;
    }
  }
}

// ---------------------------------------------------------------------------
// Top-5 per row over 50000 (reads raw logits before log_softmax transform).
// ---------------------------------------------------------------------------
__global__ __launch_bounds__(256) void topk_kernel(const float* __restrict__ lg,
                                                   int* __restrict__ kidx) {
  __shared__ float sv[256 * 5];
  __shared__ int   si[256 * 5];
  int r = blockIdx.x, tid = threadIdx.x;
  float v[5]; int ix[5];
#pragma unroll
  for (int i = 0; i < 5; ++i) { v[i] = -1e30f; ix[i] = 0; }
  for (int c = tid; c < 50000; c += 256) {
    float x = lg[(long)r * 50000 + c];
    if (x > v[4]) {
      int p = 4;
      while (p > 0 && x > v[p - 1]) { v[p] = v[p - 1]; ix[p] = ix[p - 1]; --p; }
      v[p] = x; ix[p] = c;
    }
  }
#pragma unroll
  for (int i = 0; i < 5; ++i) { sv[tid * 5 + i] = v[i]; si[tid * 5 + i] = ix[i]; }
  __syncthreads();
  for (int stride = 128; stride > 0; stride >>= 1) {
    if (tid < stride) {
      for (int i = 0; i < 5; ++i) {
        float x = sv[(tid + stride) * 5 + i];
        int   c = si[(tid + stride) * 5 + i];
        if (x > sv[tid * 5 + 4]) {
          int p = 4;
          while (p > 0 && x > sv[tid * 5 + p - 1]) {
            sv[tid * 5 + p] = sv[tid * 5 + p - 1];
            si[tid * 5 + p] = si[tid * 5 + p - 1]; --p;
          }
          sv[tid * 5 + p] = x; si[tid * 5 + p] = c;
        }
      }
    }
    __syncthreads();
  }
  if (tid < 5) kidx[r * 5 + tid] = si[tid];
}

// ---------------------------------------------------------------------------
// In-place row-wise log_softmax.
// ---------------------------------------------------------------------------
__global__ __launch_bounds__(512) void logsoftmax_kernel(float* __restrict__ lg, int N) {
  __shared__ float red[512];
  long base = (long)blockIdx.x * N;
  int tid = threadIdx.x;
  float mx = -1e30f;
  for (int c = tid; c < N; c += 512) mx = fmaxf(mx, lg[base + c]);
  red[tid] = mx; __syncthreads();
  for (int s = 256; s > 0; s >>= 1) { if (tid < s) red[tid] = fmaxf(red[tid], red[tid + s]); __syncthreads(); }
  mx = red[0]; __syncthreads();
  float sum = 0.f;
  for (int c = tid; c < N; c += 512) sum += __expf(lg[base + c] - mx);
  red[tid] = sum; __syncthreads();
  for (int s = 256; s > 0; s >>= 1) { if (tid < s) red[tid] += red[tid + s]; __syncthreads(); }
  float lse = mx + __logf(red[0]);
  for (int c = tid; c < N; c += 512) lg[base + c] -= lse;
}

// ---------------------------------------------------------------------------
// Sense scatter-softmax, in place over logits_s rows (gather first, then fill).
// ---------------------------------------------------------------------------
__global__ __launch_bounds__(256) void senses_kernel(const int* __restrict__ kidx,
                                                     const int* __restrict__ graph,
                                                     float* __restrict__ ls) {
  __shared__ float cl[160];
  __shared__ int   cd[160];
  __shared__ int   vld[160];
  __shared__ float s_mx, s_sum;
  __shared__ int   s_any;
  int r = blockIdx.x, tid = threadIdx.x;
  if (tid < 160) {
    int ki = kidx[r * 5 + tid / 32];
    int c  = graph[(long)ki * 32 + (tid & 31)];
    int vv = (c < 25000) ? 1 : 0;
    cd[tid] = c; vld[tid] = vv;
    cl[tid] = vv ? ls[(long)r * 25000 + c] : -1e9f;
  }
  __syncthreads();
  if (tid == 0) {
    float mx = -1e30f; int anyv = 0;
    for (int i = 0; i < 160; ++i) { mx = fmaxf(mx, cl[i]); anyv |= vld[i]; }
    float sum = 0.f;
    for (int i = 0; i < 160; ++i) sum += __expf(cl[i] - mx);
    s_mx = mx; s_sum = sum; s_any = anyv;
  }
  __syncthreads();
  float fill = s_any ? __logf(1e-8f) : -__logf(25000.f);
  for (int c = tid; c < 25000; c += 256) ls[(long)r * 25000 + c] = fill;
  __syncthreads();
  if (tid < 160 && vld[tid] && s_any) {
    float sm = __expf(cl[tid] - s_mx) / s_sum;
    ls[(long)r * 25000 + cd[tid]] = __logf(sm);
  }
}

// ---------------------------------------------------------------------------
// Host launch
// ---------------------------------------------------------------------------
extern "C" void kernel_launch(void* const* d_in, const int* in_sizes, int n_in,
                              void* d_out, int out_size, void* d_ws, size_t ws_size,
                              hipStream_t stream) {
  (void)in_sizes; (void)n_in; (void)out_size; (void)ws_size;
  const int*   x_indices = (const int*)d_in[0];
  const int*   graph     = (const int*)d_in[1];
  const float* X         = (const float*)d_in[2];
  const float* gat_W     = (const float*)d_in[3];
  const float* asrc      = (const float*)d_in[4];
  const float* adst      = (const float*)d_in[5];
  const float* gat_b     = (const float*)d_in[6];
  const float* m_Wih0    = (const float*)d_in[7];
  const float* m_Wih12   = (const float*)d_in[8];
  const float* m_Whh     = (const float*)d_in[9];
  const float* m_bih     = (const float*)d_in[10];
  const float* m_bhh     = (const float*)d_in[11];
  const float* s_Wih0    = (const float*)d_in[12];
  const float* s_Wih1    = (const float*)d_in[13];
  const float* s_Whh     = (const float*)d_in[14];
  const float* s_bih     = (const float*)d_in[15];
  const float* s_bhh     = (const float*)d_in[16];
  const float* W2g       = (const float*)d_in[17];
  const float* b2g       = (const float*)d_in[18];
  const float* W2s       = (const float*)d_in[19];
  const float* b2s       = (const float*)d_in[20];

  float* out_g = (float*)d_out;
  float* out_s = out_g + (long)1024 * 50000;

  char* wsp = (char*)d_ws;
  auto alloc = [&](size_t bytes) -> char* {
    char* p = wsp; wsp += (bytes + 255) & ~(size_t)255; return p;
  };
  float*     sig     = (float*)alloc((size_t)256 * 4 * 600 * 4);
  float*     bufA    = (float*)alloc((size_t)256 * 4 * 1024 * 4);
  float*     bufB    = (float*)alloc((size_t)256 * 4 * 1024 * 4);
  _Float16*  out2d_h = (_Float16*)alloc((size_t)1024 * 1024 * 2);
  _Float16*  so_h    = (_Float16*)alloc((size_t)1024 * 1024 * 2);
  int*       kidx    = (int*)alloc((size_t)1024 * 5 * 4);
  _Float16*  gatWswz = (_Float16*)alloc((size_t)19 * 10 * 512 * 2);
  const size_t WT = (size_t)3072 * 1024 * 2;
  _Float16* mW0t = (_Float16*)alloc(WT);
  _Float16* mW1t = (_Float16*)alloc(WT);
  _Float16* mW2t = (_Float16*)alloc(WT);
  _Float16* mH0t = (_Float16*)alloc(WT);
  _Float16* mH1t = (_Float16*)alloc(WT);
  _Float16* mH2t = (_Float16*)alloc(WT);
  _Float16* sW0t = (_Float16*)alloc(WT);
  _Float16* sW1t = (_Float16*)alloc(WT);
  _Float16* sH0t = (_Float16*)alloc(WT);
  _Float16* sH1t = (_Float16*)alloc(WT);

  const int SWG = (int)((3072L * 1024 + 255) / 256);
  swzT_kernel<<<SWG, 256, 0, stream>>>(m_Wih0, mW0t, 600);
  swzT_kernel<<<SWG, 256, 0, stream>>>(m_Wih12,                  mW1t, 1024);
  swzT_kernel<<<SWG, 256, 0, stream>>>(m_Wih12 + 3072L * 1024,   mW2t, 1024);
  swzT_kernel<<<SWG, 256, 0, stream>>>(m_Whh,                    mH0t, 1024);
  swzT_kernel<<<SWG, 256, 0, stream>>>(m_Whh + 3072L * 1024,     mH1t, 1024);
  swzT_kernel<<<SWG, 256, 0, stream>>>(m_Whh + 2 * 3072L * 1024, mH2t, 1024);
  swzT_kernel<<<SWG, 256, 0, stream>>>(s_Wih0, sW0t, 600);
  swzT_kernel<<<SWG, 256, 0, stream>>>(s_Wih1, sW1t, 1024);
  swzT_kernel<<<SWG, 256, 0, stream>>>(s_Whh,                    sH0t, 1024);
  swzT_kernel<<<SWG, 256, 0, stream>>>(s_Whh + 3072L * 1024,     sH1t, 1024);
  swz_gatw_kernel<<<(19 * 10 * 512 + 255) / 256, 256, 0, stream>>>(gat_W, gatWswz);

  gat_kernel<<<1024, 256, 0, stream>>>(x_indices, X, gatWswz, asrc, adst, gat_b, sig);

  // main GRU stack (3 layers)
  gru_kernel<<<1, 512, 0, stream>>>(sig, 600,   mW0t, mH0t, m_bih,        m_bhh,        bufA, nullptr);
  gru_kernel<<<1, 512, 0, stream>>>(bufA, 1024, mW1t, mH1t, m_bih + 3072, m_bhh + 3072, bufB, nullptr);
  gru_kernel<<<1, 512, 0, stream>>>(bufB, 1024, mW2t, mH2t, m_bih + 6144, m_bhh + 6144, bufA, out2d_h);
  // sense GRU stack (2 layers)
  gru_kernel<<<1, 512, 0, stream>>>(sig, 600,   sW0t, sH0t, s_bih,        s_bhh,        bufB, nullptr);
  gru_kernel<<<1, 512, 0, stream>>>(bufB, 1024, sW1t, sH1t, s_bih + 3072, s_bhh + 3072, bufA, so_h);

  // globals head: GEMM -> top-k (raw logits) -> log_softmax in place
  logits_kernel<<<dim3((50000 + 63) / 64, 8), 256, 0, stream>>>(out2d_h, W2g, b2g, out_g, 50000);
  topk_kernel<<<1024, 256, 0, stream>>>(out_g, kidx);
  logsoftmax_kernel<<<1024, 512, 0, stream>>>(out_g, 50000);

  // senses head: GEMM -> scatter softmax in place
  logits_kernel<<<dim3((25000 + 63) / 64, 8), 256, 0, stream>>>(so_h, W2s, b2s, out_s, 25000);
  senses_kernel<<<1024, 256, 0, stream>>>(kidx, graph, out_s);
}
